// GNN_node_39256001085525
// MI455X (gfx1250) — matrix operand order repaired and checked by reading, not compiled, not run on hardware
//
#include <hip/hip_runtime.h>
#include <stdint.h>

// ---- problem constants (match reference) ----
#define N_NODES 50000
#define N_EDGES 800000
#define DIM     256
#define HID     256
#define EMBW    512
#define NLAYERS 4
#define BN_EPS  1e-5f

// ---- WMMA vector types (gfx1250, wave32) ----
typedef __attribute__((ext_vector_type(16))) __bf16         v16bf;
typedef __attribute__((ext_vector_type(16))) unsigned short v16u;
typedef __attribute__((ext_vector_type(8)))  unsigned short v8u;
typedef __attribute__((ext_vector_type(8)))  float          v8f;

union Frag {
  v16u u;
  v16bf bf;
  v8u  h[2];
};

__device__ __forceinline__ unsigned short f2bf(float f) {
  // round-to-nearest-even float -> bf16
  unsigned u = __float_as_uint(f);
  unsigned r = u + 0x7FFFu + ((u >> 16) & 1u);
  return (unsigned short)(r >> 16);
}

// ---------------------------------------------------------------------------
// Weight convert + transpose: in (Lb, K, Nc) f32 row-major -> out (Lb, Nc, K) bf16
// (K-contiguous per output column so the WMMA B-fragment is a contiguous load)
// ---------------------------------------------------------------------------
__global__ __launch_bounds__(256) void k_cvt_transpose(
    const float* __restrict__ in, unsigned short* __restrict__ out,
    int Lb, int K, int Nc)
{
  size_t idx = (size_t)blockIdx.x * blockDim.x + threadIdx.x;
  size_t total = (size_t)Lb * K * Nc;
  if (idx >= total) return;
  int n = (int)(idx % Nc);
  int k = (int)((idx / Nc) % K);
  int l = (int)(idx / ((size_t)Nc * K));
  out[((size_t)l * Nc + n) * K + k] = f2bf(in[idx]);
}

__global__ __launch_bounds__(256) void k_copy_f32(
    float* __restrict__ dst, const float* __restrict__ src, size_t n)
{
  size_t i = (size_t)blockIdx.x * blockDim.x + threadIdx.x;
  if (i < n) dst[i] = src[i];
}

__global__ __launch_bounds__(256) void k_zero_f32(float* __restrict__ p, int n)
{
  int i = blockIdx.x * blockDim.x + threadIdx.x;
  if (i < n) p[i] = 0.f;
}

__global__ __launch_bounds__(256) void k_f32_to_bf16(
    const float* __restrict__ in, unsigned short* __restrict__ out, size_t n)
{
  size_t i = (size_t)blockIdx.x * blockDim.x + threadIdx.x;
  if (i < n) out[i] = f2bf(in[i]);
}

// ---------------------------------------------------------------------------
// GIN aggregation: agg[dst] += h[src] over all edges (agg pre-initialized = h).
// One thread per (edge, 4-feature chunk): float4 gather + 4 f32 atomics.
// ---------------------------------------------------------------------------
__global__ __launch_bounds__(256) void k_edge_scatter(
    const float* __restrict__ h, const int* __restrict__ ei,
    float* __restrict__ agg)
{
  size_t idx = (size_t)blockIdx.x * blockDim.x + threadIdx.x;
  if (idx >= (size_t)N_EDGES * (DIM / 4)) return;
  int e = (int)(idx >> 6);          // DIM/4 == 64 chunks per edge
  int c = (int)(idx & 63) << 2;
  int s = ei[e];
  int d = ei[N_EDGES + e];
  const float4 v = *(const float4*)(h + (size_t)s * DIM + c);
  float* o = agg + (size_t)d * DIM + c;
  atomicAdd(o + 0, v.x);
  atomicAdd(o + 1, v.y);
  atomicAdd(o + 2, v.z);
  atomicAdd(o + 3, v.w);
}

// ---------------------------------------------------------------------------
// bf16 WMMA GEMM: C(M,Nc) = A(M,K) * B(K,Nc) + bias, A row-major bf16,
// Bt = B transposed (Nc,K) bf16. Each wave owns a 16x64 output strip:
// one A fragment reused across 4 v_wmma_f32_16x16x32_bf16 per K-step.
// M must be a multiple of 16, Nc a multiple of 64, K a multiple of 32.
// ---------------------------------------------------------------------------
__global__ __launch_bounds__(256) void k_gemm_bf16(
    const unsigned short* __restrict__ A,
    const unsigned short* __restrict__ Bt,
    const float* __restrict__ bias,
    float* __restrict__ C,
    int M, int K, int Nc)
{
  const int lane = threadIdx.x & 31;
  const int gw   = blockIdx.x * (blockDim.x >> 5) + (threadIdx.x >> 5);
  const int groupsN = Nc >> 6;
  const int tileM = gw / groupsN;
  const int gN    = gw - tileM * groupsN;
  if (tileM * 16 >= M) return;   // wave-uniform: EXEC stays all-ones for WMMA

  // A fragment: lane -> row (mod 16); e=0..15 -> K=klo+e, klo = 16*(lane>=16)
  const int mA   = tileM * 16 + (lane & 15);
  const int kloA = (lane >> 4) << 4;
  // B fragment: lane -> column (mod 16); halves e -> K = e + 16*(lane>=16)
  const int n0   = (gN << 6) + (lane & 15);
  const int kloB = (lane >> 4) << 4;

  const unsigned short* Arow = A  + (size_t)mA * K + kloA;
  const unsigned short* B0   = Bt + (size_t)n0 * K + kloB;
  const size_t bstride = (size_t)16 * K;   // 16 columns apart

  v8f acc[4] = {};
  for (int k0 = 0; k0 < K; k0 += 32) {
    Frag a;
    a.h[0] = *(const v8u*)(Arow + k0);        // K = klo .. klo+7
    a.h[1] = *(const v8u*)(Arow + k0 + 8);    // K = klo+8 .. klo+15
#pragma unroll
    for (int j = 0; j < 4; ++j) {
      Frag b;
      b.u = *(const v16u*)(B0 + (size_t)j * bstride + k0);
      acc[j] = __builtin_amdgcn_wmma_f32_16x16x32_bf16(
          false, a.bf, false, b.bf, (short)0, acc[j], false, false);
    }
  }

  // D layout: element i -> row r0+i, column n0 + 16*j
  const int r0 = tileM * 16 + ((lane >> 4) << 3);
  float bb[4];
#pragma unroll
  for (int j = 0; j < 4; ++j) bb[j] = bias[n0 + 16 * j];
#pragma unroll
  for (int i = 0; i < 8; ++i) {
    float* crow = C + (size_t)(r0 + i) * Nc + n0;
#pragma unroll
    for (int j = 0; j < 4; ++j) crow[16 * j] = acc[j][i] + bb[j];
  }
}

// ---------------------------------------------------------------------------
// BatchNorm (training-mode, biased stats over node axis), two-pass.
// ---------------------------------------------------------------------------
__global__ __launch_bounds__(256) void k_colstats(
    const float* __restrict__ Z, float* __restrict__ sums,
    int M, int Nc, int rowsPer)
{
  int col = blockIdx.y * blockDim.x + threadIdx.x;
  if (col >= Nc) return;
  int r0 = blockIdx.x * rowsPer;
  int r1 = r0 + rowsPer; if (r1 > M) r1 = M;
  float s = 0.f, s2 = 0.f;
  for (int r = r0; r < r1; ++r) {
    float v = Z[(size_t)r * Nc + col];
    s += v;
    s2 = fmaf(v, v, s2);
  }
  atomicAdd(&sums[col], s);
  atomicAdd(&sums[Nc + col], s2);
}

__global__ __launch_bounds__(256) void k_finstats(
    const float* __restrict__ sums, const float* __restrict__ gamma,
    const float* __restrict__ beta, float* __restrict__ scale,
    float* __restrict__ shift, int M, int Nc)
{
  int c = blockIdx.x * blockDim.x + threadIdx.x;
  if (c >= Nc) return;
  float inv_m = 1.f / (float)M;
  float mean = sums[c] * inv_m;
  float var  = sums[Nc + c] * inv_m - mean * mean;
  float inv  = rsqrtf(var + BN_EPS);
  float sc   = gamma[c] * inv;
  scale[c] = sc;
  shift[c] = beta[c] - mean * sc;
}

// BN + ReLU + convert to bf16 (feeds GEMM2's A operand)
__global__ __launch_bounds__(256) void k_bn_relu_bf16(
    const float* __restrict__ Z, const float* __restrict__ scale,
    const float* __restrict__ shift, unsigned short* __restrict__ out,
    size_t total, int Nc)
{
  size_t i = (size_t)blockIdx.x * blockDim.x + threadIdx.x;
  if (i >= total) return;
  int c = (int)(i % Nc);
  float v = fmaf(Z[i], scale[c], shift[c]);
  v = v > 0.f ? v : 0.f;
  out[i] = f2bf(v);
}

// BN + optional ReLU, f32 out (next-layer h, or final output)
__global__ __launch_bounds__(256) void k_bn_act_f32(
    const float* __restrict__ Z, const float* __restrict__ scale,
    const float* __restrict__ shift, float* __restrict__ out,
    size_t total, int Nc, int doRelu)
{
  size_t i = (size_t)blockIdx.x * blockDim.x + threadIdx.x;
  if (i >= total) return;
  int c = (int)(i % Nc);
  float v = fmaf(Z[i], scale[c], shift[c]);
  if (doRelu && v < 0.f) v = 0.f;
  out[i] = v;
}

// ---------------------------------------------------------------------------
extern "C" void kernel_launch(void* const* d_in, const int* in_sizes, int n_in,
                              void* d_out, int out_size, void* d_ws, size_t ws_size,
                              hipStream_t stream)
{
  (void)in_sizes; (void)n_in; (void)out_size; (void)ws_size;

  const float* x   = (const float*)d_in[0];   // (N, 256)
  const int*   ei  = (const int*)  d_in[1];   // (2, E)
  const float* W1  = (const float*)d_in[2];   // (L, 256, 512)
  const float* b1  = (const float*)d_in[3];   // (L, 512)
  const float* g1  = (const float*)d_in[4];
  const float* be1 = (const float*)d_in[5];
  const float* W2  = (const float*)d_in[6];   // (L, 512, 256)
  const float* b2  = (const float*)d_in[7];   // (L, 256)
  const float* g2  = (const float*)d_in[8];
  const float* be2 = (const float*)d_in[9];
  float* out = (float*)d_out;                 // (N, 256)

  // workspace carve-out (~284 MB total)
  char*  ws  = (char*)d_ws;
  size_t off = 0;
  auto alloc = [&](size_t bytes) -> void* {
    void* p = ws + off;
    off += (bytes + 255) & ~(size_t)255;
    return p;
  };
  float*          agg  = (float*)         alloc((size_t)N_NODES * DIM  * 4); // also reused as z2
  unsigned short* aggb = (unsigned short*)alloc((size_t)N_NODES * DIM  * 2);
  float*          z1   = (float*)         alloc((size_t)N_NODES * EMBW * 4);
  unsigned short* z1b  = (unsigned short*)alloc((size_t)N_NODES * EMBW * 2);
  float*          hbuf = (float*)         alloc((size_t)N_NODES * HID  * 4);
  float*          sums = (float*)         alloc((size_t)2 * EMBW * 4);       // sum | sumsq
  float*          scl  = (float*)         alloc((size_t)EMBW * 4);
  float*          shf  = (float*)         alloc((size_t)EMBW * 4);
  unsigned short* W1t  = (unsigned short*)alloc((size_t)NLAYERS * DIM  * EMBW * 2);
  unsigned short* W2t  = (unsigned short*)alloc((size_t)NLAYERS * EMBW * HID  * 2);

  // one-time (per launch) weight bf16 convert + transpose (K-contiguous columns)
  {
    int total = NLAYERS * DIM * EMBW;  // == NLAYERS * EMBW * HID
    k_cvt_transpose<<<(total + 255) / 256, 256, 0, stream>>>(W1, W1t, NLAYERS, DIM,  EMBW);
    k_cvt_transpose<<<(total + 255) / 256, 256, 0, stream>>>(W2, W2t, NLAYERS, EMBW, HID);
  }

  const size_t nd = (size_t)N_NODES * DIM;   // 12.8M
  const size_t ne = (size_t)N_NODES * EMBW;  // 25.6M
  const size_t et = (size_t)N_EDGES * (DIM / 4);

  const float* hin = x;
  for (int l = 0; l < NLAYERS; ++l) {
    // --- GIN aggregation: agg = h + sum_{j->i} h_j ---
    k_copy_f32    <<<(int)((nd + 255) / 256), 256, 0, stream>>>(agg, hin, nd);
    k_edge_scatter<<<(int)((et + 255) / 256), 256, 0, stream>>>(hin, ei, agg);
    k_f32_to_bf16 <<<(int)((nd + 255) / 256), 256, 0, stream>>>(agg, aggb, nd);

    // --- GEMM1: (N,256) x (256,512) + b1 -> z1 ---
    {
      int waves = (N_NODES / 16) * (EMBW / 64);
      k_gemm_bf16<<<(waves + 7) / 8, 256, 0, stream>>>(
          aggb, W1t + (size_t)l * DIM * EMBW, b1 + (size_t)l * EMBW,
          z1, N_NODES, DIM, EMBW);
    }
    // --- BN1 + ReLU -> bf16 ---
    k_zero_f32<<<(2 * EMBW + 255) / 256, 256, 0, stream>>>(sums, 2 * EMBW);
    {
      dim3 g((N_NODES + 255) / 256, EMBW / 256);
      k_colstats<<<g, 256, 0, stream>>>(z1, sums, N_NODES, EMBW, 256);
    }
    k_finstats<<<(EMBW + 255) / 256, 256, 0, stream>>>(
        sums, g1 + (size_t)l * EMBW, be1 + (size_t)l * EMBW, scl, shf, N_NODES, EMBW);
    k_bn_relu_bf16<<<(int)((ne + 255) / 256), 256, 0, stream>>>(z1, scl, shf, z1b, ne, EMBW);

    // --- GEMM2: (N,512) x (512,256) + b2 -> z2 (reuse agg buffer) ---
    float* z2 = agg;
    {
      int waves = (N_NODES / 16) * (HID / 64);
      k_gemm_bf16<<<(waves + 7) / 8, 256, 0, stream>>>(
          z1b, W2t + (size_t)l * EMBW * HID, b2 + (size_t)l * HID,
          z2, N_NODES, EMBW, HID);
    }
    // --- BN2 (+ ReLU except last layer) -> f32 ---
    k_zero_f32<<<(2 * HID + 255) / 256, 256, 0, stream>>>(sums, 2 * HID);
    {
      dim3 g((N_NODES + 255) / 256, HID / 256);
      k_colstats<<<g, 256, 0, stream>>>(z2, sums, N_NODES, HID, 256);
    }
    k_finstats<<<(HID + 255) / 256, 256, 0, stream>>>(
        sums, g2 + (size_t)l * HID, be2 + (size_t)l * HID, scl, shf, N_NODES, HID);

    float* hnext = (l == NLAYERS - 1) ? out : hbuf;
    k_bn_act_f32<<<(int)((nd + 255) / 256), 256, 0, stream>>>(
        z2, scl, shf, hnext, nd, HID, (l == NLAYERS - 1) ? 0 : 1);
    hin = hnext;
  }
}
